// ChessGATv2_6571299963427
// MI455X (gfx1250) — matrix operand order repaired and checked by reading, not compiled, not run on hardware
//
#include <hip/hip_runtime.h>
#include <hip/hip_bf16.h>
#include <float.h>

// ---------------------------------------------------------------------------
// ChessGATv2 forward for MI455X (gfx1250, wave32, WMMA).
// Dense transforms use v_wmma_f32_16x16x32_f16 (f16 in / f32 accumulate).
// Weights staged TRANSPOSED in LDS so every WMMA fragment load is a
// contiguous 128-bit ds_load_b128. Elementwise/streaming kernels use b128
// global accesses; GEMM stores take a uniform full-tile fast path.
// ---------------------------------------------------------------------------

#define NN   50000
#define EE   500000
#define GG   2048
#define HIDC 128
#define NH   4
#define EDC  12
#define NL   8

typedef __attribute__((ext_vector_type(16))) _Float16 v16h;
typedef __attribute__((ext_vector_type(8)))  _Float16 v8h;
typedef __attribute__((ext_vector_type(4)))  _Float16 v4h;
typedef __attribute__((ext_vector_type(8)))  float    v8f;

// ----- monotonic float<->uint encoding for atomicMax-based segment max -----
__device__ __forceinline__ unsigned int fenc(float f) {
    unsigned int u = __float_as_uint(f);
    return (u & 0x80000000u) ? ~u : (u | 0x80000000u);
}
__device__ __forceinline__ float fdec(unsigned int v) {
    unsigned int u = (v & 0x80000000u) ? (v ^ 0x80000000u) : ~v;
    return __uint_as_float(u);
}

// ----- WMMA fragment builders (CDNA5 ISA 7.12.2 layouts) -------------------
// A (16x32 f16, row-major LDS, lda halves): lane L holds row M=L%16;
// K-halves {kb..kb+7, kb+16..kb+23}, kb = (L<16) ? 0 : 8.
__device__ __forceinline__ v16h load_a_frag(const _Float16* As, int lda,
                                            int row0, int k0, int lane) {
    int m  = lane & 15;
    int kb = (lane < 16) ? 0 : 8;
    const _Float16* p = As + (row0 + m) * lda + k0 + kb;
    union { v8h h2[2]; v16h f; } u;
    u.h2[0] = *(const v8h*)(p);
    u.h2[1] = *(const v8h*)(p + 16);
    return u.f;
}
// B (32x16 f16) from TRANSPOSED LDS weights Bt[col][k] (ldk halves):
// lane L holds col N=L%16; K-halves {kb..kb+7, kb+16..kb+23}, kb = (L<16) ? 0 : 8.
__device__ __forceinline__ v16h load_b_frag_t(const _Float16* Bt, int ldk,
                                              int k0, int col0, int lane) {
    int n  = lane & 15;
    int kb = (lane < 16) ? 0 : 8;
    const _Float16* p = Bt + (col0 + n) * ldk + k0 + kb;
    union { v8h h2[2]; v16h f; } u;
    u.h2[0] = *(const v8h*)(p);
    u.h2[1] = *(const v8h*)(p + 16);
    return u.f;
}
// C/D (16x16 f32): lane L col N=L%16; VGPR v holds row v + ((L<16)?0:8).

// ---------------------------------------------------------------------------
// h = relu(x @ Win + b_in)   [N,22]x[22,128] — tiny K, scalar FMA kernel.
// ---------------------------------------------------------------------------
__global__ __launch_bounds__(128) void input_mlp(const float* __restrict__ x,
                                                 const float* __restrict__ Win,
                                                 const float* __restrict__ bin,
                                                 float* __restrict__ h) {
    __shared__ float xs[22];
    int n = blockIdx.x, t = threadIdx.x;
    if (t < 22) xs[t] = x[n * 22 + t];
    __syncthreads();
    float s = bin[t];
#pragma unroll
    for (int k = 0; k < 22; ++k) s = fmaf(xs[k], Win[k * HIDC + t], s);
    h[n * HIDC + t] = fmaxf(s, 0.f);
}

// ---------------------------------------------------------------------------
// out = hin @ W + bias   [N,128]x[128,128], WMMA. Block = 64-row stripe.
// LDS: h stripe (16KB f16, row-major) + weight (32KB f16, TRANSPOSED).
// float4 staging, A-fragments hoisted, uniform full-tile store fast path.
// ---------------------------------------------------------------------------
__global__ __launch_bounds__(256) void gemm_node(const float* __restrict__ hin,
                                                 const float* __restrict__ W,
                                                 const float* __restrict__ bias,
                                                 float* __restrict__ out) {
    __shared__ __align__(32) _Float16 hs[64 * HIDC];
    __shared__ __align__(32) _Float16 Wt[HIDC * HIDC];   // [col][k]
    int tid  = threadIdx.x;
    int row0 = blockIdx.x * 64;
    bool full = (row0 + 64 <= NN);
    // stage weight transposed: b128 global reads, 2B scattered LDS writes
    const float4* W4 = (const float4*)W;
    for (int i = tid; i < HIDC * HIDC / 4; i += 256) {
        int k = i >> 5, c4 = (i & 31) * 4;
        float4 v = W4[i];
        Wt[(c4 + 0) * HIDC + k] = (_Float16)v.x;
        Wt[(c4 + 1) * HIDC + k] = (_Float16)v.y;
        Wt[(c4 + 2) * HIDC + k] = (_Float16)v.z;
        Wt[(c4 + 3) * HIDC + k] = (_Float16)v.w;
    }
    // stage h stripe: b128 global reads, b64 LDS writes
    const float4* hin4 = (const float4*)hin;
    for (int i = tid; i < 64 * HIDC / 4; i += 256) {
        int r = i >> 5, c4 = i & 31;
        int gr = row0 + r;
        float4 v = (gr < NN) ? hin4[gr * 32 + c4] : make_float4(0.f, 0.f, 0.f, 0.f);
        v4h hv; hv[0] = (_Float16)v.x; hv[1] = (_Float16)v.y;
        hv[2] = (_Float16)v.z; hv[3] = (_Float16)v.w;
        *(v4h*)(hs + r * HIDC + c4 * 4) = hv;
    }
    __syncthreads();
    int wave = tid >> 5, lane = tid & 31;
    int rt    = wave & 3;          // 4 row tiles of 16
    int cbase = (wave >> 2) * 4;   // 8 col tiles split over 2 wave groups
    v16h afr[4];
#pragma unroll
    for (int kk = 0; kk < 4; ++kk)
        afr[kk] = load_a_frag(hs, HIDC, rt * 16, kk * 32, lane);
    int n = lane & 15, mb = (lane < 16) ? 0 : 8;
#pragma unroll
    for (int ci = 0; ci < 4; ++ci) {
        int ct = cbase + ci;
        v8f acc = {};
#pragma unroll
        for (int kk = 0; kk < 4; ++kk) {
            v16h b = load_b_frag_t(Wt, HIDC, kk * 32, ct * 16, lane);
            acc = __builtin_amdgcn_wmma_f32_16x16x32_f16(false, afr[kk], false, b,
                                                         (short)0, acc, false, false);
        }
        int col = ct * 16 + n;
        float bv = bias[col];
        int gr0 = row0 + rt * 16 + mb;
        if (full) {
#pragma unroll
            for (int v = 0; v < 8; ++v)
                out[(gr0 + v) * HIDC + col] = acc[v] + bv;
        } else {
#pragma unroll
            for (int v = 0; v < 8; ++v)
                if (gr0 + v < NN) out[(gr0 + v) * HIDC + col] = acc[v] + bv;
        }
    }
}

// ---------------------------------------------------------------------------
// Per-layer init: zero aggregation (b128), denominators, BN accumulators;
// seg-max initialized to encoded -FLT_MAX.
// ---------------------------------------------------------------------------
__global__ void layer_init(float4* agg4, float* den, unsigned int* mseg,
                           float* colsum, float* colsq) {
    int idx = blockIdx.x * 256 + threadIdx.x;
    if (idx < NN * HIDC / 4) agg4[idx] = make_float4(0.f, 0.f, 0.f, 0.f);
    if (idx < NN * NH) { den[idx] = 0.f; mseg[idx] = fenc(-FLT_MAX); }
    if (idx < HIDC) { colsum[idx] = 0.f; colsq[idx] = 0.f; }
}

// ---------------------------------------------------------------------------
// Pass 1: per-edge attention logits a[e,h] + segment max over dst.
// One wave per edge; lane handles cols {h*32+lane}. Edge-feature projection
// (12x128) computed on the fly from LDS-staged We. Butterfly reduce over the
// 32 lanes == reduction over head dim D=32.
// ---------------------------------------------------------------------------
__global__ __launch_bounds__(256) void edge_score(const float* __restrict__ xl,
                                                  const float* __restrict__ xr,
                                                  const float* __restrict__ ea,
                                                  const int* __restrict__ src,
                                                  const int* __restrict__ dst,
                                                  const float* __restrict__ We,
                                                  const float* __restrict__ att,
                                                  float* __restrict__ aeh,
                                                  unsigned int* __restrict__ mseg) {
    __shared__ float wes[EDC * HIDC];
    __shared__ float atts[HIDC];
    int tid = threadIdx.x;
    for (int i = tid; i < EDC * HIDC; i += 256) wes[i] = We[i];
    if (tid < HIDC) atts[tid] = att[tid];
    __syncthreads();
    int wave = tid >> 5, lane = tid & 31;
    int e = blockIdx.x * 8 + wave;
    if (e >= EE) return;
    int s = src[e], d = dst[e];
    float ev[EDC];
#pragma unroll
    for (int k = 0; k < EDC; ++k) ev[k] = ea[e * EDC + k];
    float ph[NH];
#pragma unroll
    for (int hh = 0; hh < NH; ++hh) {
        int c = hh * 32 + lane;
        float v = xl[s * HIDC + c] + xr[d * HIDC + c];
#pragma unroll
        for (int k = 0; k < EDC; ++k) v = fmaf(ev[k], wes[k * HIDC + c], v);
        v = (v > 0.f) ? v : 0.2f * v;             // LeakyReLU(0.2)
        ph[hh] = v * atts[c];                      // att[hh][lane]
    }
#pragma unroll
    for (int off = 16; off > 0; off >>= 1) {
#pragma unroll
        for (int hh = 0; hh < NH; ++hh) ph[hh] += __shfl_xor(ph[hh], off, 32);
    }
    if (lane == 0) {
#pragma unroll
        for (int hh = 0; hh < NH; ++hh) {
            aeh[e * NH + hh] = ph[hh];
            atomicMax(&mseg[d * NH + hh], fenc(ph[hh]));
        }
    }
}

// Pass 2: ex = exp(a - m[dst]); accumulate denominators; overwrite aeh.
__global__ void edge_soft(float* __restrict__ aeh,
                          const unsigned int* __restrict__ mseg,
                          float* __restrict__ den,
                          const int* __restrict__ dst) {
    int i = blockIdx.x * 256 + threadIdx.x;
    if (i >= EE * NH) return;
    int e = i >> 2, hh = i & 3;
    int d = dst[e];
    float ex = __expf(aeh[i] - fdec(mseg[d * NH + hh]));
    aeh[i] = ex;
    atomicAdd(&den[d * NH + hh], ex);
}

// Pass 3: alpha = ex/den[dst]; agg[dst] += xl[src] * alpha  (wave per edge).
__global__ __launch_bounds__(256) void edge_aggr(const float* __restrict__ xl,
                                                 const float* __restrict__ aeh,
                                                 const float* __restrict__ den,
                                                 const int* __restrict__ src,
                                                 const int* __restrict__ dst,
                                                 float* __restrict__ agg) {
    int wave = threadIdx.x >> 5, lane = threadIdx.x & 31;
    int e = blockIdx.x * 8 + wave;
    if (e >= EE) return;
    int s = src[e], d = dst[e];
    float al[NH];
#pragma unroll
    for (int hh = 0; hh < NH; ++hh)
        al[hh] = aeh[e * NH + hh] / (den[d * NH + hh] + 1e-16f);
#pragma unroll
    for (int hh = 0; hh < NH; ++hh) {
        int c = hh * 32 + lane;
        atomicAdd(&agg[d * HIDC + c], xl[s * HIDC + c] * al[hh]);
    }
}

// ---------------------------------------------------------------------------
// BatchNorm: two-pass column reduction over N rows.
// Note: GAT bias bo is dropped — BN(x + const_per_col) == BN(x).
// ---------------------------------------------------------------------------
__global__ __launch_bounds__(256) void bn_stats(const float* __restrict__ agg,
                                                float* __restrict__ colsum,
                                                float* __restrict__ colsq) {
    __shared__ float ss[256], sq[256];
    int col = threadIdx.x & 127, half = threadIdx.x >> 7;
    int r0 = blockIdx.x * 200;
    int r1 = r0 + 200; if (r1 > NN) r1 = NN;
    float s = 0.f, q = 0.f;
    for (int r = r0 + half; r < r1; r += 2) {
        float v = agg[r * HIDC + col];
        s += v; q += v * v;
    }
    ss[threadIdx.x] = s; sq[threadIdx.x] = q;
    __syncthreads();
    if (half == 0) {
        atomicAdd(&colsum[col], ss[col] + ss[col + 128]);
        atomicAdd(&colsq[col],  sq[col] + sq[col + 128]);
    }
}

__global__ void bn_final(const float* colsum, const float* colsq,
                         float* mu, float* rs) {
    int c = threadIdx.x;
    float m = colsum[c] * (1.f / (float)NN);
    float v = colsq[c] * (1.f / (float)NN) - m * m;
    mu[c] = m;
    rs[c] = rsqrtf(fmaxf(v, 0.f) + 1e-5f);
}

// mode 0: h = relu(bn)     (even layer)
// mode 1: h = relu(bn+res) (odd layer, residual)
// b128 streaming: one float4 (4 consecutive cols) per thread.
__global__ void bn_apply(const float4* __restrict__ agg4,
                         const float* __restrict__ mu, const float* __restrict__ rs,
                         const float* __restrict__ gamma, const float* __restrict__ beta,
                         const float4* __restrict__ res4, float4* __restrict__ hout4,
                         int add_res) {
    int idx = blockIdx.x * 256 + threadIdx.x;
    if (idx >= NN * HIDC / 4) return;
    int c = (idx & 31) * 4;
    float4 a = agg4[idx];
    float4 o;
    o.x = (a.x - mu[c + 0]) * rs[c + 0] * gamma[c + 0] + beta[c + 0];
    o.y = (a.y - mu[c + 1]) * rs[c + 1] * gamma[c + 1] + beta[c + 1];
    o.z = (a.z - mu[c + 2]) * rs[c + 2] * gamma[c + 2] + beta[c + 2];
    o.w = (a.w - mu[c + 3]) * rs[c + 3] * gamma[c + 3] + beta[c + 3];
    if (add_res) {
        float4 r = res4[idx];
        o.x += r.x; o.y += r.y; o.z += r.z; o.w += r.w;
    }
    o.x = fmaxf(o.x, 0.f); o.y = fmaxf(o.y, 0.f);
    o.z = fmaxf(o.z, 0.f); o.w = fmaxf(o.w, 0.f);
    hout4[idx] = o;
}

// ---------------------------------------------------------------------------
// Graph pooling + value head.
// ---------------------------------------------------------------------------
__global__ void pool_init(float* pooled, float* cnt) {
    int idx = blockIdx.x * 256 + threadIdx.x;
    if (idx < GG * HIDC) pooled[idx] = 0.f;
    if (idx < GG) cnt[idx] = 0.f;
}

__global__ void pool_scatter(const float* __restrict__ h,
                             const int* __restrict__ batch,
                             float* __restrict__ pooled, float* __restrict__ cnt) {
    int idx = blockIdx.x * 256 + threadIdx.x;
    if (idx >= NN * HIDC) return;
    int n = idx >> 7, c = idx & 127;
    int g = batch[n];
    atomicAdd(&pooled[g * HIDC + c], h[idx]);
    if (c == 0) atomicAdd(&cnt[g], 1.f);
}

__global__ __launch_bounds__(64) void value_head(const float* __restrict__ pooled,
                                                 const float* __restrict__ cnt,
                                                 const float* __restrict__ Wv1,
                                                 const float* __restrict__ bv1,
                                                 const float* __restrict__ Wv2,
                                                 const float* __restrict__ bv2,
                                                 float* __restrict__ out) {
    __shared__ float ps[HIDC];
    __shared__ float hid[64];
    int g = blockIdx.x, t = threadIdx.x;
    float inv = 1.f / fmaxf(cnt[g], 1.f);
    ps[t]      = pooled[g * HIDC + t] * inv;
    ps[t + 64] = pooled[g * HIDC + 64 + t] * inv;
    __syncthreads();
    float s = bv1[t];
#pragma unroll 8
    for (int k = 0; k < HIDC; ++k) s = fmaf(ps[k], Wv1[k * 64 + t], s);
    hid[t] = fmaxf(s, 0.f);
    __syncthreads();
    if (t < 3) {
        float o = bv2[t];
#pragma unroll 8
        for (int k = 0; k < 64; ++k) o = fmaf(hid[k], Wv2[k * 3 + t], o);
        out[g * 3 + t] = o;
    }
}

// ---------------------------------------------------------------------------
// Policy head (WMMA): rows = [h[src] | h[dst] | edge_attr] padded to K=288.
// Block = 32 edges; LDS: A 18KB (row-major) + W 36KB (f16, TRANSPOSED)
// + hidden 8KB (f32). 9 WMMAs per wave, then 64-wide dot with Wp2.
// ---------------------------------------------------------------------------
__global__ __launch_bounds__(256) void policy_head(const float* __restrict__ h,
                                                   const int* __restrict__ src,
                                                   const int* __restrict__ dst,
                                                   const float* __restrict__ ea,
                                                   const float* __restrict__ Wp1,
                                                   const float* __restrict__ bp1,
                                                   const float* __restrict__ Wp2,
                                                   const float* __restrict__ bp2,
                                                   float* __restrict__ pol) {
    __shared__ __align__(32) _Float16 As[32 * 288];
    __shared__ __align__(32) _Float16 Wt[64 * 288];   // [col][k], k padded to 288
    __shared__ float Hs[32 * 64];
    int tid = threadIdx.x;
    int e0  = blockIdx.x * 32;
    for (int i = tid; i < 288 * 64; i += 256) {
        int k = i >> 6, c = i & 63;
        Wt[c * 288 + k] = (_Float16)((k < 268) ? Wp1[k * 64 + c] : 0.f);
    }
    for (int i = tid; i < 32 * 288; i += 256) {
        int s = i / 288, k = i - s * 288;
        int e = e0 + s;
        float v = 0.f;
        if (e < EE) {
            if (k < 128)      v = h[src[e] * HIDC + k];
            else if (k < 256) v = h[dst[e] * HIDC + (k - 128)];
            else if (k < 268) v = ea[e * EDC + (k - 256)];
        }
        As[i] = (_Float16)v;
    }
    __syncthreads();
    int wave = tid >> 5, lane = tid & 31;
    int rt = wave >> 2;   // 2 edge tiles of 16
    int ct = wave & 3;    // 4 col tiles of 16
    v8f acc = {};
#pragma unroll
    for (int kk = 0; kk < 9; ++kk) {
        v16h a = load_a_frag(As, 288, rt * 16, kk * 32, lane);
        v16h b = load_b_frag_t(Wt, 288, kk * 32, ct * 16, lane);
        acc = __builtin_amdgcn_wmma_f32_16x16x32_f16(false, a, false, b,
                                                     (short)0, acc, false, false);
    }
    int n = lane & 15, mb = (lane < 16) ? 0 : 8;
    int col = ct * 16 + n;
    float bv = bp1[col];
#pragma unroll
    for (int v = 0; v < 8; ++v) {
        float hv = acc[v] + bv;
        Hs[(rt * 16 + mb + v) * 64 + col] = (hv > 0.f) ? hv : 0.f;
    }
    __syncthreads();
    if (tid < 32) {
        int e = e0 + tid;
        if (e < EE) {
            float s = bp2[0];
#pragma unroll 8
            for (int k = 0; k < 64; ++k) s = fmaf(Hs[tid * 64 + k], Wp2[k], s);
            pol[e] = s;
        }
    }
}

// ---------------------------------------------------------------------------
// Workspace layout (bytes, all 256-aligned). ~139 MB total.
// ---------------------------------------------------------------------------
static constexpr size_t SZ_NODE  = (size_t)NN * HIDC * sizeof(float);
static constexpr size_t OFF_H    = 0;
static constexpr size_t OFF_RES  = OFF_H   + SZ_NODE;
static constexpr size_t OFF_AGG  = OFF_RES + SZ_NODE;
static constexpr size_t OFF_XL   = OFF_AGG + SZ_NODE;
static constexpr size_t OFF_XR   = OFF_XL  + SZ_NODE;
static constexpr size_t OFF_AEH  = OFF_XR  + SZ_NODE;                 // E*4 f32
static constexpr size_t OFF_MSEG = OFF_AEH + (size_t)EE * NH * 4;     // N*4 u32
static constexpr size_t OFF_DEN  = OFF_MSEG + (size_t)NN * NH * 4;    // N*4 f32
static constexpr size_t OFF_CS   = OFF_DEN + (size_t)NN * NH * 4;
static constexpr size_t OFF_CQ   = OFF_CS  + 512;
static constexpr size_t OFF_MU   = OFF_CQ  + 512;
static constexpr size_t OFF_RSQ  = OFF_MU  + 512;
static constexpr size_t OFF_PL   = OFF_RSQ + 512;                     // G*128 f32
static constexpr size_t OFF_CNT  = OFF_PL  + (size_t)GG * HIDC * 4;   // G f32

extern "C" void kernel_launch(void* const* d_in, const int* in_sizes, int n_in,
                              void* d_out, int out_size, void* d_ws, size_t ws_size,
                              hipStream_t stream) {
    const float* x     = (const float*)d_in[0];
    const int*   src   = (const int*)  d_in[1];
    const int*   dst   = (const int*)  d_in[2];
    const int*   batch = (const int*)  d_in[3];
    const float* ea    = (const float*)d_in[4];
    const float* Win   = (const float*)d_in[5];
    const float* b_in  = (const float*)d_in[6];
    const float* Wl    = (const float*)d_in[7];
    const float* bl    = (const float*)d_in[8];
    const float* Wr    = (const float*)d_in[9];
    const float* br    = (const float*)d_in[10];
    const float* We    = (const float*)d_in[11];
    const float* att   = (const float*)d_in[12];
    // d_in[13] = bo: skipped, BatchNorm cancels per-column constants.
    const float* gamma = (const float*)d_in[14];
    const float* beta  = (const float*)d_in[15];
    const float* Wv1   = (const float*)d_in[16];
    const float* bv1   = (const float*)d_in[17];
    const float* Wv2   = (const float*)d_in[18];
    const float* bv2   = (const float*)d_in[19];
    const float* Wp1   = (const float*)d_in[20];
    const float* bp1   = (const float*)d_in[21];
    const float* Wp2   = (const float*)d_in[22];
    const float* bp2   = (const float*)d_in[23];

    char* w = (char*)d_ws;
    float*        h      = (float*)(w + OFF_H);
    float*        res    = (float*)(w + OFF_RES);
    float*        agg    = (float*)(w + OFF_AGG);
    float*        xl     = (float*)(w + OFF_XL);
    float*        xr     = (float*)(w + OFF_XR);
    float*        aeh    = (float*)(w + OFF_AEH);
    unsigned int* mseg   = (unsigned int*)(w + OFF_MSEG);
    float*        den    = (float*)(w + OFF_DEN);
    float*        colsum = (float*)(w + OFF_CS);
    float*        colsq  = (float*)(w + OFF_CQ);
    float*        mu     = (float*)(w + OFF_MU);
    float*        rs     = (float*)(w + OFF_RSQ);
    float*        pooled = (float*)(w + OFF_PL);
    float*        cnt    = (float*)(w + OFF_CNT);

    float* out_val = (float*)d_out;            // [G,3]
    float* out_pol = (float*)d_out + GG * 3;   // [E]

    const int elemGrid  = (NN * HIDC + 255) / 256;       // scalar elementwise
    const int vec4Grid  = (NN * HIDC / 4 + 255) / 256;   // float4 streaming
    const int gemmGrid  = (NN + 63) / 64;                // 782
    const int edgeGrid  = EE / 8;                        // 62500

    input_mlp<<<NN, 128, 0, stream>>>(x, Win, b_in, h);

    for (int l = 0; l < NL; ++l) {
        if ((l & 1) == 0)
            hipMemcpyAsync(res, h, SZ_NODE, hipMemcpyDeviceToDevice, stream);
        layer_init<<<vec4Grid, 256, 0, stream>>>((float4*)agg, den, mseg, colsum, colsq);
        gemm_node<<<gemmGrid, 256, 0, stream>>>(h, Wl + l * HIDC * HIDC, bl + l * HIDC, xl);
        gemm_node<<<gemmGrid, 256, 0, stream>>>(h, Wr + l * HIDC * HIDC, br + l * HIDC, xr);
        edge_score<<<edgeGrid, 256, 0, stream>>>(xl, xr, ea, src, dst,
                                                 We + l * EDC * HIDC, att + l * NH * 32,
                                                 aeh, mseg);
        edge_soft<<<(EE * NH + 255) / 256, 256, 0, stream>>>(aeh, mseg, den, dst);
        edge_aggr<<<edgeGrid, 256, 0, stream>>>(xl, aeh, den, src, dst, agg);
        bn_stats<<<(NN + 199) / 200, 256, 0, stream>>>(agg, colsum, colsq);
        bn_final<<<1, 128, 0, stream>>>(colsum, colsq, mu, rs);
        bn_apply<<<vec4Grid, 256, 0, stream>>>((const float4*)agg, mu, rs,
                                               gamma + l * HIDC, beta + l * HIDC,
                                               (const float4*)res, (float4*)h, (l & 1));
    }

    pool_init<<<(GG * HIDC + 255) / 256, 256, 0, stream>>>(pooled, cnt);
    pool_scatter<<<elemGrid, 256, 0, stream>>>(h, batch, pooled, cnt);
    value_head<<<GG, 64, 0, stream>>>(pooled, cnt, Wv1, bv1, Wv2, bv2, out_val);
    policy_head<<<(EE + 31) / 32, 256, 0, stream>>>(h, src, dst, ea,
                                                    Wp1, bp1, Wp2, bp2, out_pol);
}
